// GATModel_54168127537296
// MI455X (gfx1250) — matrix operand order repaired and checked by reading, not compiled, not run on hardware
//
#include <hip/hip_runtime.h>
#include <hip/hip_bf16.h>
#include <math.h>

// ---- problem constants (match reference) ----
#define NNODES 50000
#define NEDGES 500000
#define ETOT   (NEDGES + NNODES)   // with self-loops
#define FIN    128
#define HID1   64
#define NHEAD  8
#define F1     (NHEAD * HID1)      // 512
#define NOUT   40
#define NOUTP  48                  // padded to 3 x 16 WMMA col tiles
#define NEG_SLOPE 0.2f

typedef __attribute__((ext_vector_type(16))) __bf16 v16bf;
typedef __attribute__((ext_vector_type(8)))  float  v8f;

struct ushort4x { unsigned short a, b, c, d; };

// ---------------- small device helpers ----------------
__device__ __forceinline__ unsigned short f2bf(float f) {
  unsigned u = __float_as_uint(f);
  u += 0x7FFFu + ((u >> 16) & 1u);   // round-to-nearest-even
  return (unsigned short)(u >> 16);
}
__device__ __forceinline__ float lrelu(float x) { return x > 0.f ? x : NEG_SLOPE * x; }

__device__ __forceinline__ void atomicMaxF(float* addr, float val) {
  unsigned* ua = (unsigned*)addr;
  unsigned old = *ua;
  while (__uint_as_float(old) < val) {
    unsigned assumed = old;
    old = atomicCAS(ua, assumed, __float_as_uint(val));
    if (old == assumed) break;
  }
}

// ---------------- utility kernels ----------------
__global__ void fill_f32(float* __restrict__ p, float v, int n) {
  int i = blockIdx.x * blockDim.x + threadIdx.x;
  if (i < n) p[i] = v;
}

// vector fill: n must be a multiple of 4
__global__ void fill_f32x4(float4* __restrict__ p, float v, int n4) {
  int i = blockIdx.x * blockDim.x + threadIdx.x;
  if (i < n4) p[i] = make_float4(v, v, v, v);
}

// vector convert f32 -> bf16: n must be a multiple of 4
__global__ void cvt_f32_bf16x4(const float4* __restrict__ src, ushort4x* __restrict__ dst, int n4) {
  int i = blockIdx.x * blockDim.x + threadIdx.x;
  if (i >= n4) return;
  float4 f = src[i];
  ushort4x o;
  o.a = f2bf(f.x); o.b = f2bf(f.y); o.c = f2bf(f.z); o.d = f2bf(f.w);
  dst[i] = o;
}

// W1 [128,512] row-major -> W1T bf16 [512,128]
__global__ void transpose_w1(const float* __restrict__ W, unsigned short* __restrict__ WT) {
  int t = blockIdx.x * blockDim.x + threadIdx.x;
  if (t >= F1 * FIN) return;
  int nn = t / FIN, k = t - nn * FIN;
  WT[t] = f2bf(W[k * F1 + nn]);
}

// W2 [512,40] row-major -> W2T bf16 [48,512], zero-padded cols 40..47
__global__ void transpose_w2(const float* __restrict__ W, unsigned short* __restrict__ WT) {
  int t = blockIdx.x * blockDim.x + threadIdx.x;
  if (t >= NOUTP * F1) return;
  int nn = t / F1, k = t - nn * F1;
  WT[t] = (nn < NOUT) ? f2bf(W[k * NOUT + nn]) : (unsigned short)0;
}

// ---------------- WMMA GEMM: C[M,ldc] = A[M,K](bf16) x B[K,N](bf16, given as BT[N,K]) ----------------
// One wave computes one 16x16 f32 output tile. K looped in steps of 32, fully unrolled.
template <int KDIM>
__global__ void wmma_gemm_bf16(const unsigned short* __restrict__ A,
                               const unsigned short* __restrict__ BT,
                               float* __restrict__ C,
                               int colTiles, int ldc, int totalTiles) {
  int wave = (blockIdx.x * blockDim.x + threadIdx.x) >> 5;
  if (wave >= totalTiles) return;               // whole-wave uniform; EXEC stays all-ones for WMMA
  int lane = threadIdx.x & 31;
  int rowTile = wave / colTiles;
  int colTile = wave - rowTile * colTiles;
  int lm = lane & 15;
  int hi = lane >> 4;

  const unsigned* Arow = (const unsigned*)(A + (size_t)(rowTile * 16 + lm) * KDIM);
  const unsigned* Brow = (const unsigned*)(BT + (size_t)(colTile * 16 + lm) * KDIM);
  const int aHi = hi * 8;    // A: hi half-lanes take K+8 (and K+24) per ISA layout
  const int bHi = hi * 16;   // B: uses same layout as A (aHi)

  v8f c = {};
#pragma unroll
  for (int k0 = 0; k0 < KDIM; k0 += 32) {
    union { v16bf v; unsigned u[8]; } a, b;
#pragma unroll
    for (int r = 0; r < 8; ++r) {
      int kA = k0 + ((r & 4) ? 16 : 0) + aHi + (r & 3) * 2;  // even -> aligned 32-bit pair
      a.u[r] = Arow[kA >> 1];
      int kB = k0 + ((r & 4) ? 16 : 0) + aHi + (r & 3) * 2;
      b.u[r] = Brow[kB >> 1];
    }
    c = __builtin_amdgcn_wmma_f32_16x16x32_bf16(false, a.v, false, b.v, (short)0, c, false, false);
  }

  int col = colTile * 16 + lm;
  int rbase = rowTile * 16 + hi * 8;
#pragma unroll
  for (int r = 0; r < 8; ++r)
    C[(size_t)(rbase + r) * ldc + col] = c[r];
}

// ---------------- attention scalar kernels, layer 1 ----------------
// e_src/e_dst per (node, head): dot(h1[n, h*64 : h*64+64], a1_*[h])
__global__ void node_e1(const float* __restrict__ h1, const float* __restrict__ a1s,
                        const float* __restrict__ a1d, float* __restrict__ es,
                        float* __restrict__ ed) {
  int t = blockIdx.x * blockDim.x + threadIdx.x;
  if (t >= NNODES * NHEAD) return;
  int n = t >> 3, h = t & 7;
  const float4* hp = (const float4*)(h1 + (size_t)n * F1 + h * HID1);
  const float4* as = (const float4*)(a1s + h * HID1);
  const float4* ad = (const float4*)(a1d + h * HID1);
  float ss = 0.f, sd = 0.f;
#pragma unroll
  for (int ch = 0; ch < HID1 / 4; ++ch) {
    float4 x = hp[ch], s4 = as[ch], d4 = ad[ch];
    ss += x.x * s4.x + x.y * s4.y + x.z * s4.z + x.w * s4.w;
    sd += x.x * d4.x + x.y * d4.y + x.z * d4.z + x.w * d4.w;
  }
  es[t] = ss; ed[t] = sd;
}

__global__ void edge_max1(const int* __restrict__ ei, const float* __restrict__ es,
                          const float* __restrict__ ed, float* __restrict__ mx) {
  int t = blockIdx.x * blockDim.x + threadIdx.x;
  if (t >= ETOT * NHEAD) return;
  int e = t >> 3, h = t & 7;
  int s = (e < NEDGES) ? ei[e] : (e - NEDGES);
  int d = (e < NEDGES) ? ei[NEDGES + e] : (e - NEDGES);
  float lg = lrelu(es[s * NHEAD + h] + ed[d * NHEAD + h]);
  atomicMaxF(&mx[d * NHEAD + h], lg);
}

__global__ void edge_sum1(const int* __restrict__ ei, const float* __restrict__ es,
                          const float* __restrict__ ed, const float* __restrict__ mx,
                          float* __restrict__ sm, float* __restrict__ al) {
  int t = blockIdx.x * blockDim.x + threadIdx.x;
  if (t >= ETOT * NHEAD) return;
  int e = t >> 3, h = t & 7;
  int s = (e < NEDGES) ? ei[e] : (e - NEDGES);
  int d = (e < NEDGES) ? ei[NEDGES + e] : (e - NEDGES);
  float lg = lrelu(es[s * NHEAD + h] + ed[d * NHEAD + h]);
  float p = __expf(lg - mx[d * NHEAD + h]);
  al[t] = p;
  atomicAdd(&sm[d * NHEAD + h], p);
}

// out[dst, h*64+c..c+3] += h1[src, h*64+c..c+3] * alpha(e,h)
// one thread per (edge, 4 channels): float4 gather + 4 f32 atomics
__global__ void edge_agg1(const int* __restrict__ ei, const float* __restrict__ al,
                          const float* __restrict__ sm, const float* __restrict__ h1,
                          float* __restrict__ ag) {
  long long t = (long long)blockIdx.x * blockDim.x + threadIdx.x;
  if (t >= (long long)ETOT * (F1 / 4)) return;
  int e = (int)(t >> 7);               // F1/4 == 128 quads per edge
  int q = (int)(t & (F1 / 4 - 1));
  int r = q << 2;                      // channel base, multiple of 4
  int h = r >> 6;                      // HID1 == 64
  int s = (e < NEDGES) ? ei[e] : (e - NEDGES);
  int d = (e < NEDGES) ? ei[NEDGES + e] : (e - NEDGES);
  const float4* hp = (const float4*)(h1 + (size_t)s * F1 + r);
  __builtin_prefetch(hp, 0, 1);
  float alpha = al[e * NHEAD + h] / (sm[d * NHEAD + h] + 1e-16f);
  float4 x = *hp;
  float* o = ag + (size_t)d * F1 + r;
  atomicAdd(o + 0, x.x * alpha);
  atomicAdd(o + 1, x.y * alpha);
  atomicAdd(o + 2, x.z * alpha);
  atomicAdd(o + 3, x.w * alpha);
}

// layer1 epilogue: h2 = bf16(relu(agg + b1)), vectorized by 4
__global__ void h2prep(const float4* __restrict__ ag, const float4* __restrict__ b1,
                       ushort4x* __restrict__ h2b) {
  int t = blockIdx.x * blockDim.x + threadIdx.x;
  if (t >= NNODES * (F1 / 4)) return;
  int c4 = t & (F1 / 4 - 1);
  float4 v = ag[t], b = b1[c4];
  float vx = v.x + b.x, vy = v.y + b.y, vz = v.z + b.z, vw = v.w + b.w;
  ushort4x o;
  o.a = f2bf(vx > 0.f ? vx : 0.f);
  o.b = f2bf(vy > 0.f ? vy : 0.f);
  o.c = f2bf(vz > 0.f ? vz : 0.f);
  o.d = f2bf(vw > 0.f ? vw : 0.f);
  h2b[t] = o;
}

// ---------------- attention scalar kernels, layer 2 (1 head, 40 ch, s2 stride 48) ----------------
__global__ void node_e2(const float* __restrict__ s2, const float* __restrict__ a2s,
                        const float* __restrict__ a2d, float* __restrict__ es,
                        float* __restrict__ ed) {
  int n = blockIdx.x * blockDim.x + threadIdx.x;
  if (n >= NNODES) return;
  const float* hp = s2 + (size_t)n * NOUTP;
  float ss = 0.f, sd = 0.f;
#pragma unroll
  for (int c = 0; c < NOUT; ++c) { float x = hp[c]; ss += x * a2s[c]; sd += x * a2d[c]; }
  es[n] = ss; ed[n] = sd;
}

__global__ void edge_max2(const int* __restrict__ ei, const float* __restrict__ es,
                          const float* __restrict__ ed, float* __restrict__ mx) {
  int e = blockIdx.x * blockDim.x + threadIdx.x;
  if (e >= ETOT) return;
  int s = (e < NEDGES) ? ei[e] : (e - NEDGES);
  int d = (e < NEDGES) ? ei[NEDGES + e] : (e - NEDGES);
  atomicMaxF(&mx[d], lrelu(es[s] + ed[d]));
}

__global__ void edge_sum2(const int* __restrict__ ei, const float* __restrict__ es,
                          const float* __restrict__ ed, const float* __restrict__ mx,
                          float* __restrict__ sm, float* __restrict__ al) {
  int e = blockIdx.x * blockDim.x + threadIdx.x;
  if (e >= ETOT) return;
  int s = (e < NEDGES) ? ei[e] : (e - NEDGES);
  int d = (e < NEDGES) ? ei[NEDGES + e] : (e - NEDGES);
  float p = __expf(lrelu(es[s] + ed[d]) - mx[d]);
  al[e] = p;
  atomicAdd(&sm[d], p);
}

// one thread per (edge, 4 channels): 10 quads cover 40 outputs
__global__ void edge_agg2(const int* __restrict__ ei, const float* __restrict__ al,
                          const float* __restrict__ sm, const float* __restrict__ s2,
                          float* __restrict__ out) {
  int t = blockIdx.x * blockDim.x + threadIdx.x;
  if (t >= ETOT * (NOUT / 4)) return;
  int e = t / (NOUT / 4);
  int q = t - e * (NOUT / 4);
  int c0 = q << 2;
  int s = (e < NEDGES) ? ei[e] : (e - NEDGES);
  int d = (e < NEDGES) ? ei[NEDGES + e] : (e - NEDGES);
  float alpha = al[e] / (sm[d] + 1e-16f);
  float4 x = *(const float4*)(s2 + (size_t)s * NOUTP + c0);
  float* o = out + (size_t)d * NOUT + c0;
  atomicAdd(o + 0, x.x * alpha);
  atomicAdd(o + 1, x.y * alpha);
  atomicAdd(o + 2, x.z * alpha);
  atomicAdd(o + 3, x.w * alpha);
}

// add b2 and log_softmax in place
__global__ void finalize(float* __restrict__ out, const float* __restrict__ b2) {
  int n = blockIdx.x * blockDim.x + threadIdx.x;
  if (n >= NNODES) return;
  float v[NOUT];
  float m = -INFINITY;
#pragma unroll
  for (int c = 0; c < NOUT; ++c) { v[c] = out[(size_t)n * NOUT + c] + b2[c]; m = fmaxf(m, v[c]); }
  float sum = 0.f;
#pragma unroll
  for (int c = 0; c < NOUT; ++c) sum += __expf(v[c] - m);
  float lse = __logf(sum);
#pragma unroll
  for (int c = 0; c < NOUT; ++c) out[(size_t)n * NOUT + c] = v[c] - m - lse;
}

// ---------------- host side ----------------
extern "C" void kernel_launch(void* const* d_in, const int* in_sizes, int n_in,
                              void* d_out, int out_size, void* d_ws, size_t ws_size,
                              hipStream_t stream) {
  const float* x   = (const float*)d_in[0];
  const int*   ei  = (const int*)  d_in[1];
  const float* W1  = (const float*)d_in[2];
  const float* a1s = (const float*)d_in[3];
  const float* a1d = (const float*)d_in[4];
  const float* b1  = (const float*)d_in[5];
  const float* W2  = (const float*)d_in[6];
  const float* a2s = (const float*)d_in[7];
  const float* a2d = (const float*)d_in[8];
  const float* b2  = (const float*)d_in[9];
  float* out = (float*)d_out;

  // carve workspace (256B-aligned slices)
  char* p = (char*)d_ws;
  auto carve = [&](size_t bytes) -> void* {
    void* r = (void*)p;
    p += (bytes + 255) & ~(size_t)255;
    return r;
  };
  unsigned short* Xb  = (unsigned short*)carve((size_t)NNODES * FIN * 2);
  unsigned short* W1T = (unsigned short*)carve((size_t)F1 * FIN * 2);
  float* h1  = (float*)carve((size_t)NNODES * F1 * 4);
  float* es1 = (float*)carve((size_t)NNODES * NHEAD * 4);
  float* ed1 = (float*)carve((size_t)NNODES * NHEAD * 4);
  float* mx1 = (float*)carve((size_t)NNODES * NHEAD * 4);
  float* sm1 = (float*)carve((size_t)NNODES * NHEAD * 4);
  float* al1 = (float*)carve((size_t)ETOT * NHEAD * 4);
  float* ag1 = (float*)carve((size_t)NNODES * F1 * 4);
  unsigned short* h2b = (unsigned short*)carve((size_t)NNODES * F1 * 2);
  unsigned short* W2T = (unsigned short*)carve((size_t)NOUTP * F1 * 2);
  float* s2  = (float*)carve((size_t)NNODES * NOUTP * 4);
  float* es2 = (float*)carve((size_t)NNODES * 4);
  float* ed2 = (float*)carve((size_t)NNODES * 4);
  float* mx2 = (float*)carve((size_t)NNODES * 4);
  float* sm2 = (float*)carve((size_t)NNODES * 4);
  float* al2 = (float*)carve((size_t)ETOT * 4);

  const int T = 256;
  auto nb = [](long long n, int t) { return (int)((n + t - 1) / t); };

  // ---- layer 1 ----
  cvt_f32_bf16x4<<<nb((long long)NNODES * FIN / 4, T), T, 0, stream>>>(
      (const float4*)x, (ushort4x*)Xb, NNODES * FIN / 4);
  transpose_w1<<<nb((long long)F1 * FIN, T), T, 0, stream>>>(W1, W1T);

  // GEMM1: h1[N,512] = Xb[N,128] @ W1 ; 3125*32 = 100000 wave tiles, 8 waves/block
  wmma_gemm_bf16<FIN><<<100000 / 8, T, 0, stream>>>(Xb, W1T, h1, F1 / 16, F1, 100000);

  node_e1<<<nb((long long)NNODES * NHEAD, T), T, 0, stream>>>(h1, a1s, a1d, es1, ed1);

  fill_f32<<<nb((long long)NNODES * NHEAD, T), T, 0, stream>>>(mx1, -INFINITY, NNODES * NHEAD);
  fill_f32<<<nb((long long)NNODES * NHEAD, T), T, 0, stream>>>(sm1, 0.f, NNODES * NHEAD);
  fill_f32x4<<<nb((long long)NNODES * F1 / 4, T), T, 0, stream>>>((float4*)ag1, 0.f, NNODES * F1 / 4);

  edge_max1<<<nb((long long)ETOT * NHEAD, T), T, 0, stream>>>(ei, es1, ed1, mx1);
  edge_sum1<<<nb((long long)ETOT * NHEAD, T), T, 0, stream>>>(ei, es1, ed1, mx1, sm1, al1);
  edge_agg1<<<nb((long long)ETOT * (F1 / 4), T), T, 0, stream>>>(ei, al1, sm1, h1, ag1);

  h2prep<<<nb((long long)NNODES * F1 / 4, T), T, 0, stream>>>(
      (const float4*)ag1, (const float4*)b1, (ushort4x*)h2b);

  // ---- layer 2 ----
  transpose_w2<<<nb((long long)NOUTP * F1, T), T, 0, stream>>>(W2, W2T);

  // GEMM2: s2[N,48] = h2b[N,512] @ W2pad ; 3125*3 = 9375 wave tiles
  wmma_gemm_bf16<F1><<<nb((long long)9375 * 32, T), T, 0, stream>>>(h2b, W2T, s2, NOUTP / 16, NOUTP, 9375);

  node_e2<<<nb(NNODES, T), T, 0, stream>>>(s2, a2s, a2d, es2, ed2);

  fill_f32<<<nb(NNODES, T), T, 0, stream>>>(mx2, -INFINITY, NNODES);
  fill_f32<<<nb(NNODES, T), T, 0, stream>>>(sm2, 0.f, NNODES);
  fill_f32x4<<<nb((long long)NNODES * NOUT / 4, T), T, 0, stream>>>((float4*)out, 0.f, NNODES * NOUT / 4);

  edge_max2<<<nb(ETOT, T), T, 0, stream>>>(ei, es2, ed2, mx2);
  edge_sum2<<<nb(ETOT, T), T, 0, stream>>>(ei, es2, ed2, mx2, sm2, al2);
  edge_agg2<<<nb((long long)ETOT * (NOUT / 4), T), T, 0, stream>>>(ei, al2, sm2, s2, out);

  finalize<<<nb(NNODES, T), T, 0, stream>>>(out, b2);
  (void)in_sizes; (void)n_in; (void)out_size; (void)ws_size;
}
